// GRUDecoder_25262997636033
// MI455X (gfx1250) — matrix operand order repaired and checked by reading, not compiled, not run on hardware
//
#include <hip/hip_runtime.h>

// ---------------------------------------------------------------------------
// Problem constants (from reference): B=128, T=128, S=256, E=256, H=512
// ---------------------------------------------------------------------------
#define BB   128
#define TT   128
#define SS   256
#define EE   256
#define HH   512
#define NEGV (-1000000000.0f)

typedef __attribute__((ext_vector_type(8)))  float  v8f;
typedef __attribute__((ext_vector_type(16))) __bf16 v16bf;

// ---- bf16 helpers (RNE) ----------------------------------------------------
__device__ __host__ inline __bf16 f2bf(float f) {
  unsigned u = __builtin_bit_cast(unsigned, f);
  u += 0x7FFFu + ((u >> 16) & 1u);
  unsigned short s = (unsigned short)(u >> 16);
  return __builtin_bit_cast(__bf16, s);
}

// ---- WMMA fragment helpers -------------------------------------------------
// A-matrix 16x32 bf16 from LDS, documented layout:
//   lane<16: row m=lane, K = {kb..kb+7, kb+16..kb+23}, kb=0
//   lane>=16: same row m=lane-15? no: m=lane&15, kb=8
// consecutive-k pairs -> one 32-bit LDS read per pair.
__device__ inline v16bf load_a_frag(const __bf16* row_base, int row_stride, int kbase) {
  int lane = threadIdx.x & 31;
  int m  = lane & 15;
  int kb = kbase + ((lane >> 4) << 3);
  const unsigned int* p = (const unsigned int*)(row_base + (size_t)m * row_stride);
  union { v16bf v; unsigned int u[8]; } af;
#pragma unroll
  for (int j = 0; j < 8; ++j) {
    int k0 = kb + ((j < 4) ? (2 * j) : (2 * j + 8));
    af.u[j] = p[k0 >> 1];
  }
  return af.v;
}

// B-matrix 32x16 bf16 from fragment-swizzled global weights:
// block = (ktile*Ntiles + ntile), 512 elts; lane's 16 values contiguous.
__device__ inline v16bf load_b_frag(const __bf16* wsw, int blk) {
  int lane = threadIdx.x & 31;
  return *(const v16bf*)(wsw + (size_t)blk * 512 + lane * 16);
}

__device__ inline v8f wmma_bf16(v16bf a, v16bf b, v8f c) {
  return __builtin_amdgcn_wmma_f32_16x16x32_bf16(false, a, false, b, (short)0, c,
                                                 false, false);
}

// ---------------------------------------------------------------------------
// Weight swizzle: f32 (optionally transposed) -> fragment-ordered bf16.
// dst element o: blk=o>>9, l=(o&511)>>4, t=o&15; k=kt*32+((l>>4)<<3)+(t<8?t:t+8); n=nt*16+(l&15)
// ---------------------------------------------------------------------------
__global__ void swizzle_weights(const float* __restrict__ src, __bf16* __restrict__ dst,
                                int K, int N, int transpose) {
  int total = K * N;
  int ntiles = N >> 4;
  for (int o = blockIdx.x * blockDim.x + threadIdx.x; o < total;
       o += gridDim.x * blockDim.x) {
    int blk = o >> 9, r = o & 511;
    int l = r >> 4, t = r & 15;
    int kt = blk / ntiles, nt = blk - kt * ntiles;
    int k = kt * 32 + ((l >> 4) << 3) + ((t < 8) ? t : t + 8);
    int n = nt * 16 + (l & 15);
    float v = transpose ? src[(size_t)n * K + k] : src[(size_t)k * N + n];
    dst[o] = f2bf(v);
  }
}

// ---------------------------------------------------------------------------
// proj_key = encoder_hidden(B*S,H) @ W_key(H,H)  -- full-GPU WMMA GEMM
// block: 256 threads (8 waves), Mtile=16, Ntile=128 (8 waves x 16), 4 N-groups
// ---------------------------------------------------------------------------
__global__ void __launch_bounds__(256) proj_key_kernel(const float* __restrict__ eh,
                                                       const __bf16* __restrict__ wkey,
                                                       float* __restrict__ pk) {
  __shared__ __bf16 As[16 * HH];  // 16 KB staging, bf16 A tile
  int mt = blockIdx.x >> 2;
  int ng = blockIdx.x & 3;
  int row0 = mt * 16;
  for (int i = threadIdx.x; i < 16 * HH; i += blockDim.x)
    As[i] = f2bf(eh[(size_t)(row0 + (i >> 9)) * HH + (i & 511)]);
  __syncthreads();

  int w = threadIdx.x >> 5;
  int nt = ng * 8 + w;  // 0..31 over N=512
  v8f acc = {};
#pragma unroll
  for (int kt = 0; kt < HH / 32; ++kt) {
    v16bf a = load_a_frag(As, HH, kt * 32);
    v16bf b = load_b_frag(wkey, kt * 32 + nt);
    acc = wmma_bf16(a, b, acc);
  }
  int lane = threadIdx.x & 31;
  int n = nt * 16 + (lane & 15);
  int m0 = (lane >> 4) << 3;
#pragma unroll
  for (int j = 0; j < 8; ++j) pk[(size_t)(row0 + m0 + j) * HH + n] = acc[j];
}

// ---------------------------------------------------------------------------
// Persistent GRU+attention scan. 8 blocks (one per 16-batch slice), 1024 thr.
// LDS carve (305152 B < 320 KB WGP LDS):
//   gi  f32 [16][1536]   (also reused as q [16][512] between GEMM phases)
//   gh  f32 [16][1536]
//   hS  f32 [16][512]
//   sc  f32 [16][256]    (scores -> alphas)
//   vE  f32 [512]
//   act bf16 [16][1280]  = [x(256) | h_new(512) | ctx(512)]
//   hB  bf16 [16][512]
// ---------------------------------------------------------------------------
__global__ void __launch_bounds__(1024, 1) gru_main(
    const float* __restrict__ trg_embed, const float* __restrict__ eh,
    const float* __restrict__ enc_final, const unsigned char* __restrict__ src_mask,
    const float* __restrict__ v_energy, const float* __restrict__ b_ih,
    const float* __restrict__ b_hh, const __bf16* __restrict__ wq,
    const __bf16* __restrict__ wih, const __bf16* __restrict__ whh,
    const __bf16* __restrict__ wpre, const float* __restrict__ pk,
    float* __restrict__ out_ds, float* __restrict__ out_hidden,
    float* __restrict__ out_pre) {
  extern __shared__ char smem[];
  float*  gi  = (float*)smem;        // 24576 f
  float*  gh  = gi + 16 * 1536;      // 24576 f
  float*  hS  = gh + 16 * 1536;      // 8192 f
  float*  sc  = hS + 16 * 512;       // 4096 f
  float*  vE  = sc + 16 * 256;       // 512 f
  __bf16* act = (__bf16*)(vE + 512); // 20480 bf16
  __bf16* hB  = act + 16 * 1280;     // 8192 bf16

  const int tid  = threadIdx.x;
  const int w    = tid >> 5;
  const int lane = tid & 31;
  const int mb0  = blockIdx.x * 16;  // batch slice base

  // ---- init: v_energy, h0 = encoder_final ----
  for (int i = tid; i < HH; i += 1024) vE[i] = v_energy[i];
  for (int p = 0; p < 8; ++p) {
    int idx = tid + p * 1024;           // 16*512
    int b = idx >> 9, j = idx & 511;
    float hv = enc_final[(size_t)(mb0 + b) * HH + j];
    hS[idx] = hv;
    hB[idx] = f2bf(hv);
  }
  __syncthreads();

  for (int t = 0; t < TT; ++t) {
    // ---- phase 0: stage x_t (bf16) into act[:,0:256] ----
    for (int p = 0; p < 4; ++p) {
      int idx = tid + p * 1024;         // 16*256
      int b = idx >> 8, c = idx & 255;
      const float* xp = trg_embed + ((size_t)(mb0 + b) * TT + t) * EE + c;
      act[b * 1280 + c] = f2bf(*xp);
      if (t + 1 < TT) __builtin_prefetch(xp + EE, 0, 0);
    }

    // ---- phase 1: q = h @ W_query (M16,K512,N512), D -> gi (reused as q) ----
    {
      v8f acc = {};
#pragma unroll
      for (int kt = 0; kt < 16; ++kt) {
        v16bf a = load_a_frag(hB, HH, kt * 32);
        v16bf b = load_b_frag(wq, kt * 32 + w);  // Ntiles=32, nt=w
        acc = wmma_bf16(a, b, acc);
      }
      int n = lane & 15, m0 = (lane >> 4) << 3;
#pragma unroll
      for (int j = 0; j < 8; ++j) gi[(m0 + j) * 512 + w * 16 + n] = acc[j];
    }
    __syncthreads();

    // ---- phase 2: energies  sc[b][s] = sum_h v[h]*tanh(q+pk)  (masked) ----
    for (int p = 0; p < 4; ++p) {
      int idx = tid + p * 1024;         // 16*256 pairs
      int b = idx >> 8, s = idx & 255;
      const float4* qp  = (const float4*)(gi + b * 512);
      const float4* pp  = (const float4*)(pk + ((size_t)(mb0 + b) * SS + s) * HH);
      const float4* vp  = (const float4*)vE;
      float acc = 0.f;
      for (int h4 = 0; h4 < HH / 4; ++h4) {
        float4 q4 = qp[h4], p4 = pp[h4], v4 = vp[h4];
        acc += v4.x * tanhf(q4.x + p4.x) + v4.y * tanhf(q4.y + p4.y) +
               v4.z * tanhf(q4.z + p4.z) + v4.w * tanhf(q4.w + p4.w);
      }
      sc[b * 256 + s] = src_mask[(size_t)(mb0 + b) * SS + s] ? acc : NEGV;
    }
    __syncthreads();

    // ---- phase 3: softmax over S per row (wave32 shuffles, waves 0..15) ----
    if (w < 16) {
      float vals[8], vmax = -3.4e38f;
#pragma unroll
      for (int k = 0; k < 8; ++k) {
        vals[k] = sc[w * 256 + k * 32 + lane];
        vmax = fmaxf(vmax, vals[k]);
      }
#pragma unroll
      for (int off = 16; off; off >>= 1) vmax = fmaxf(vmax, __shfl_xor(vmax, off, 32));
      float ssum = 0.f;
#pragma unroll
      for (int k = 0; k < 8; ++k) { vals[k] = __expf(vals[k] - vmax); ssum += vals[k]; }
#pragma unroll
      for (int off = 16; off; off >>= 1) ssum += __shfl_xor(ssum, off, 32);
      float inv = 1.f / ssum;
#pragma unroll
      for (int k = 0; k < 8; ++k) sc[w * 256 + k * 32 + lane] = vals[k] * inv;
    }
    __syncthreads();

    // ---- phase 4: context = alphas @ encoder_hidden -> act[:,768:1280] ----
    for (int p = 0; p < 8; ++p) {
      int idx = tid + p * 1024;         // 16*512
      int b = idx >> 9, h = idx & 511;
      const float* al = sc + b * 256;
      const float* ep = eh + (size_t)(mb0 + b) * SS * HH + h;
      float acc = 0.f;
      for (int s = 0; s < SS; ++s) acc += al[s] * ep[(size_t)s * HH];
      act[b * 1280 + 768 + h] = f2bf(acc);
    }
    __syncthreads();

    // ---- phase 5: gi = [x,ctx] @ W_ih^T  (K=768, N=1536, 3 ntiles/wave) ----
    {
      v8f a0 = {}, a1 = {}, a2 = {};
#pragma unroll
      for (int kt = 0; kt < 24; ++kt) {
        int col = (kt < 8) ? kt * 32 : 768 + (kt - 8) * 32;
        v16bf a = load_a_frag(act, 1280, col);
        a0 = wmma_bf16(a, load_b_frag(wih, kt * 96 + w * 3 + 0), a0);
        a1 = wmma_bf16(a, load_b_frag(wih, kt * 96 + w * 3 + 1), a1);
        a2 = wmma_bf16(a, load_b_frag(wih, kt * 96 + w * 3 + 2), a2);
      }
      int n = lane & 15, m0 = (lane >> 4) << 3;
#pragma unroll
      for (int j = 0; j < 8; ++j) {
        gi[(m0 + j) * 1536 + (w * 3 + 0) * 16 + n] = a0[j];
        gi[(m0 + j) * 1536 + (w * 3 + 1) * 16 + n] = a1[j];
        gi[(m0 + j) * 1536 + (w * 3 + 2) * 16 + n] = a2[j];
      }
    }
    // ---- phase 6: gh = h @ W_hh^T (K=512, N=1536) ----
    {
      v8f a0 = {}, a1 = {}, a2 = {};
#pragma unroll
      for (int kt = 0; kt < 16; ++kt) {
        v16bf a = load_a_frag(hB, HH, kt * 32);
        a0 = wmma_bf16(a, load_b_frag(whh, kt * 96 + w * 3 + 0), a0);
        a1 = wmma_bf16(a, load_b_frag(whh, kt * 96 + w * 3 + 1), a1);
        a2 = wmma_bf16(a, load_b_frag(whh, kt * 96 + w * 3 + 2), a2);
      }
      int n = lane & 15, m0 = (lane >> 4) << 3;
#pragma unroll
      for (int j = 0; j < 8; ++j) {
        gh[(m0 + j) * 1536 + (w * 3 + 0) * 16 + n] = a0[j];
        gh[(m0 + j) * 1536 + (w * 3 + 1) * 16 + n] = a1[j];
        gh[(m0 + j) * 1536 + (w * 3 + 2) * 16 + n] = a2[j];
      }
    }
    __syncthreads();

    // ---- phase 7: GRU gates, h update, write decoder_states ----
    float hnew_reg[8];
    for (int p = 0; p < 8; ++p) {
      int idx = tid + p * 1024;         // 16*512
      int b = idx >> 9, j = idx & 511;
      float ir = gi[b * 1536 + j]        + b_ih[j];
      float iz = gi[b * 1536 + 512 + j]  + b_ih[512 + j];
      float in_ = gi[b * 1536 + 1024 + j] + b_ih[1024 + j];
      float hr = gh[b * 1536 + j]        + b_hh[j];
      float hz = gh[b * 1536 + 512 + j]  + b_hh[512 + j];
      float hn = gh[b * 1536 + 1024 + j] + b_hh[1024 + j];
      float r = 1.f / (1.f + __expf(-(ir + hr)));
      float z = 1.f / (1.f + __expf(-(iz + hz)));
      float n = tanhf(in_ + r * hn);
      float hv = (1.f - z) * n + z * hS[b * 512 + j];
      hnew_reg[p] = hv;
      out_ds[((size_t)(mb0 + b) * TT + t) * HH + j] = hv;
    }
    __syncthreads();
    for (int p = 0; p < 8; ++p) {
      int idx = tid + p * 1024;
      int b = idx >> 9, j = idx & 511;
      float hv = hnew_reg[p];
      hS[idx] = hv;
      hB[idx] = f2bf(hv);
      act[b * 1280 + 256 + j] = f2bf(hv);
    }
    __syncthreads();

    // ---- phase 8: pre = [x,h_new,ctx] @ W_pre (K=1280, N=512) -> global ----
    {
      v8f acc = {};
#pragma unroll
      for (int kt = 0; kt < 40; ++kt) {
        v16bf a = load_a_frag(act, 1280, kt * 32);
        acc = wmma_bf16(a, load_b_frag(wpre, kt * 32 + w), acc);
      }
      int n = lane & 15, m0 = (lane >> 4) << 3;
#pragma unroll
      for (int j = 0; j < 8; ++j)
        out_pre[((size_t)(mb0 + m0 + j) * TT + t) * HH + w * 16 + n] = acc[j];
    }
    __syncthreads();  // protect act/hB before next iteration's writes
  }

  // ---- final hidden state ----
  for (int p = 0; p < 8; ++p) {
    int idx = tid + p * 1024;
    int b = idx >> 9, j = idx & 511;
    out_hidden[(size_t)(mb0 + b) * HH + j] = hS[idx];
  }
}

// ---------------------------------------------------------------------------
// Workspace layout (element offsets):
//   bf16: Wq[262144] | WihT[1179648] | WhhT[786432] | Wpre[655360] | Wkey[262144]
//   then f32 proj_key[B*S*H]  -> total ~73.4 MB
// ---------------------------------------------------------------------------
#define WQ_OFF   0
#define WIH_OFF  262144
#define WHH_OFF  1441792
#define WPRE_OFF 2228224
#define WKEY_OFF 2883584
#define PK_BYTE  6291456

extern "C" void kernel_launch(void* const* d_in, const int* in_sizes, int n_in,
                              void* d_out, int out_size, void* d_ws, size_t ws_size,
                              hipStream_t stream) {
  (void)in_sizes; (void)n_in; (void)out_size; (void)ws_size;
  const float* trg_embed = (const float*)d_in[0];
  const float* eh        = (const float*)d_in[1];
  const float* enc_final = (const float*)d_in[2];
  const unsigned char* src_mask = (const unsigned char*)d_in[3];
  /* d_in[4] trg_mask unused by reference */
  const float* W_key   = (const float*)d_in[5];
  const float* W_query = (const float*)d_in[6];
  const float* v_ener  = (const float*)d_in[7];
  const float* W_ih    = (const float*)d_in[8];
  const float* W_hh    = (const float*)d_in[9];
  const float* b_ih    = (const float*)d_in[10];
  const float* b_hh    = (const float*)d_in[11];
  const float* W_pre   = (const float*)d_in[12];

  __bf16* wsb  = (__bf16*)d_ws;
  __bf16* wq   = wsb + WQ_OFF;
  __bf16* wih  = wsb + WIH_OFF;
  __bf16* whh  = wsb + WHH_OFF;
  __bf16* wpre = wsb + WPRE_OFF;
  __bf16* wkey = wsb + WKEY_OFF;
  float*  pk   = (float*)((char*)d_ws + PK_BYTE);

  float* out_ds     = (float*)d_out;
  float* out_hidden = out_ds + (size_t)BB * TT * HH;
  float* out_pre    = out_hidden + (size_t)BB * HH;

  // 1) weight conversion into fragment-swizzled bf16
  swizzle_weights<<<1024, 256, 0, stream>>>(W_query, wq, 512, 512, 0);
  swizzle_weights<<<4608, 256, 0, stream>>>(W_ih, wih, 768, 1536, 1);
  swizzle_weights<<<3072, 256, 0, stream>>>(W_hh, whh, 512, 1536, 1);
  swizzle_weights<<<2560, 256, 0, stream>>>(W_pre, wpre, 1280, 512, 0);
  swizzle_weights<<<1024, 256, 0, stream>>>(W_key, wkey, 512, 512, 0);

  // 2) proj_key GEMM: (B*S/16)*4 = 8192 blocks
  proj_key_kernel<<<8192, 256, 0, stream>>>(eh, wkey, pk);

  // 3) persistent batch-sliced GRU scan (8 blocks, 298 KB LDS each)
  size_t smem = 305152;
  hipFuncSetAttribute((const void*)gru_main,
                      hipFuncAttributeMaxDynamicSharedMemorySize, (int)smem);
  gru_main<<<BB / 16, 1024, smem, stream>>>(
      trg_embed, eh, enc_final, src_mask, v_ener, b_ih, b_hh, wq, wih, whh, wpre,
      pk, out_ds, out_hidden, out_pre);
}
